// SOPBC_23201413333270
// MI455X (gfx1250) — matrix operand order repaired and checked by reading, not compiled, not run on hardware
//
#include <hip/hip_runtime.h>

typedef __attribute__((ext_vector_type(16))) _Float16 v16h;
typedef __attribute__((ext_vector_type(8)))  float    v8f;

#define T_LEN   16384
#define NB      2
#define NMN     449
#define NK      11
#define LHALO   50
#define TT      128                 // t-columns per workgroup (8 waves x 16)
#define NCHUNK  15                  // ceil(449/32) K-chunks of 32
#define WIN     (TT + 2*LHALO)      // E window = 228
#define AROW    40                  // LDS row stride in halves (80B, 16B aligned)

#define WS_CPACK_OFF 4096           // after mn table (449*8=3592B)

// ---------------- prep 1: build (m,n) index table in reference order -------
__global__ void build_mn(int2* mn) {
    if (threadIdx.x == 0 && blockIdx.x == 0) {
        int idx = 0;
        for (int m = -25; m <= 25; ++m)
            for (int n = -25; n <= 25; ++n) {
                int p = m * n; if (p < 0) p = -p;
                if (p <= 25) { mn[idx].x = m; mn[idx].y = n; ++idx; }
            }
    }
}

// ---------------- prep 2: pack coefficients into WMMA A-fragment images ----
// cpack linear index = (((b*4+tile)*NCHUNK+chunk)*32+lane)*8 + dword
// tile0: C1_real rows k=0..10, row11 = pbc_C_real   (x Ar / x Ai)
// tile1: C1_imag rows k=0..10, row11 = pbc_C_imag
// tile2: C2_real rows k=0..10
// tile3: C2_imag rows k=0..10
__global__ void pack_coefs(const float* __restrict__ task_info,
                           const float* __restrict__ pbcCr, const float* __restrict__ pbcCi,
                           const float* __restrict__ C1r,  const float* __restrict__ C1i,
                           const float* __restrict__ C2r,  const float* __restrict__ C2i,
                           unsigned int* __restrict__ cpack) {
    int g = blockIdx.x * blockDim.x + threadIdx.x;
    const int total = NB * 4 * NCHUNK * 32 * 8;
    if (g >= total) return;
    int r = g;
    int d    = r & 7;   r >>= 3;
    int lane = r & 31;  r >>= 5;
    int chunk = r % NCHUNK; r /= NCHUNK;
    int tile = r & 3;
    int b    = r >> 2;

    // select coefficient set per batch (argmax of x == Rs)
    int x = (int)(task_info[b * 4 + 2] / 2.0e9f);
    int ind = 0;
    if (x == 160) ind = 3;
    if (x == 80)  ind = 2;
    if (x == 40)  ind = 1;
    if (x == 20)  ind = 0;

    int M  = lane & 15;                                        // output row
    int kb = ((lane < 16) ? 0 : 16) + 2 * d;

    unsigned int packed = 0;
    #pragma unroll
    for (int h = 0; h < 2; ++h) {
        int K  = kb + h;
        int mg = chunk * 32 + K;                               // global m index
        float v = 0.0f;
        if (mg < NMN) {
            if (tile == 0) {
                if (M <= 10)      v = C1r[ind * (NMN * NK) + mg * NK + M];
                else if (M == 11) v = pbcCr[ind * NMN + mg];
            } else if (tile == 1) {
                if (M <= 10)      v = C1i[ind * (NMN * NK) + mg * NK + M];
                else if (M == 11) v = pbcCi[ind * NMN + mg];
            } else if (tile == 2) {
                if (M <= 10)      v = C2r[ind * (NMN * NK) + mg * NK + M];
            } else {
                if (M <= 10)      v = C2i[ind * (NMN * NK) + mg * NK + M];
            }
        }
        union { _Float16 f; unsigned short u; } cv;
        cv.f = (_Float16)v;
        packed |= ((unsigned int)cv.u) << (16 * h);
    }
    cpack[g] = packed;
}

// ---------------- main fused kernel ---------------------------------------
__global__ __launch_bounds__(256)
void sopbc_main(const float* __restrict__ Er, const float* __restrict__ Ei,
                const float* __restrict__ task_info,
                const int2* __restrict__ mn,
                const unsigned int* __restrict__ cpack,
                float* __restrict__ out) {
    __shared__ float2 Ew[WIN];
    __shared__ __align__(16) _Float16 Al[2][TT][AROW];   // Ar/Ai chunk, [t][m]
    __shared__ float4 red[256];

    const int nTB = T_LEN / TT;
    int blk = blockIdx.x;
    int b   = blk / nTB;
    int t0  = (blk % nTB) * TT;
    int tid  = threadIdx.x;
    int lane = tid & 31;
    int wave = tid >> 5;

    // stage E window [t0-50, t0+TT+50) with wraparound (T is pow2)
    for (int j = tid; j < WIN; j += 256) {
        int tg = (t0 - LHALO + j) & (T_LEN - 1);
        Ew[j] = make_float2(Er[b * T_LEN + tg], Ei[b * T_LEN + tg]);
    }

    v8f acc[8];
    #pragma unroll
    for (int i = 0; i < 8; ++i) acc[i] = (v8f){0,0,0,0,0,0,0,0};

    const int tl = wave * 16 + (lane & 15);   // this lane's t column (local)
    const int kb = (lane < 16) ? 0 : 16;      // B-fragment K base

    for (int chunk = 0; chunk < NCHUNK; ++chunk) {
        __syncthreads();   // previous WMMA reads of Al done; Ew ready (iter 0)
        // regenerate A = E[t-n]*conj(E[t-m-n])*E[t-m] for this 32-row K-chunk
        for (int idx = tid; idx < 32 * TT; idx += 256) {
            int mL = idx & 31;
            int t  = idx >> 5;
            float ar = 0.0f, ai = 0.0f;
            int mg = chunk * 32 + mL;
            if (mg < NMN) {
                int2 p = mn[mg];                  // p.x = m, p.y = n
                int j = t + LHALO;
                float2 e1 = Ew[j - p.y];
                float2 e2 = Ew[j - p.x - p.y];
                float2 e3 = Ew[j - p.x];
                float cr = e1.x * e2.x + e1.y * e2.y;   // e1*conj(e2)
                float ci = e1.y * e2.x - e1.x * e2.y;
                ar = cr * e3.x - ci * e3.y;
                ai = cr * e3.y + ci * e3.x;
            }
            Al[0][t][mL] = (_Float16)ar;
            Al[1][t][mL] = (_Float16)ai;
        }
        __syncthreads();

        // coefficient fragments (prepacked per-lane images, 2x b128 each)
        v16h cf[4];
        #pragma unroll
        for (int tile = 0; tile < 4; ++tile) {
            const uint4* p = (const uint4*)(cpack +
                (size_t)((((b * 4 + tile) * NCHUNK + chunk) * 32 + lane) * 8));
            union { uint4 u[2]; v16h h; } U;
            U.u[0] = p[0]; U.u[1] = p[1];
            cf[tile] = U.h;
        }
        // signal B fragments: 16 consecutive K-halves at this lane's t column
        v16h br = *(const v16h*)&Al[0][tl][kb];
        v16h bi = *(const v16h*)&Al[1][tl][kb];

        #pragma unroll
        for (int tile = 0; tile < 4; ++tile) {
            acc[tile * 2 + 0] = __builtin_amdgcn_wmma_f32_16x16x32_f16(
                false, cf[tile], false, br, (short)0, acc[tile * 2 + 0], false, false);
            acc[tile * 2 + 1] = __builtin_amdgcn_wmma_f32_16x16x32_f16(
                false, cf[tile], false, bi, (short)0, acc[tile * 2 + 1], false, false);
        }
    }

    // epilogue: complex recombination + k-weighted partial reduction
    // acc: [0]=C1r*Ar [1]=C1r*Ai [2]=C1i*Ar [3]=C1i*Ai
    //      [4]=C2r*Ar [5]=C2r*Ai [6]=C2i*Ar [7]=C2i*Ai
    const int jt = tl + LHALO;
    const bool hi = (lane >= 16);
    float pr = 0.0f, pi = 0.0f, s1r = 0.0f, s1i = 0.0f;
    #pragma unroll
    for (int r = 0; r < 8; ++r) {
        float g1r = acc[0][r] - acc[3][r];
        float g1i = acc[1][r] + acc[2][r];
        float g2r = acc[4][r] + acc[7][r];
        float g2i = acc[6][r] - acc[5][r];
        int M = r + (hi ? 8 : 0);
        if (M <= 10) {
            float2 e = Ew[jt - (M - 5)];          // shift ks[M] = M-5
            float I  = e.x * e.x + e.y * e.y;     // |E|^2
            float Qr = e.x * e.x - e.y * e.y;     // Re(E^2)
            float Qi = 2.0f * e.x * e.y;          // Im(E^2)
            pr += I * g1r + Qr * g2r - Qi * g2i;
            pi += I * g1i + Qr * g2i + Qi * g2r;
        } else if (M == 11) {                     // first-order term row
            s1r = g1r; s1i = g1i;
        }
    }
    red[tid] = make_float4(pr, pi, s1r, s1i);
    __syncthreads();

    if (lane < 16) {
        float4 a0 = red[tid];
        float4 a1 = red[tid + 16];                // rows 8..15 half (incl. M=11)
        int tg = t0 + tl;
        if (tg >= LHALO && tg < T_LEN - LHALO) {
            float P   = exp10f(task_info[b * 4 + 0] * 0.1f);
            float s   = sqrtf(P);
            float sP3 = s * s * s;
            float sP5 = sP3 * P;
            float2 e  = Ew[tl + LHALO];
            float outr = e.x + sP3 * a1.z + sP5 * (a0.x + a1.x);
            float outi = e.y + sP3 * a1.w + sP5 * (a0.y + a1.y);
            size_t base = (size_t)(b * (T_LEN - 2 * LHALO) + (tg - LHALO)) * 2;
            out[base + 0] = outr;
            out[base + 1] = outi;
        }
    }
}

extern "C" void kernel_launch(void* const* d_in, const int* in_sizes, int n_in,
                              void* d_out, int out_size, void* d_ws, size_t ws_size,
                              hipStream_t stream) {
    (void)in_sizes; (void)n_in; (void)out_size; (void)ws_size;
    const float* Er    = (const float*)d_in[0];
    const float* Ei    = (const float*)d_in[1];
    const float* task  = (const float*)d_in[2];
    const float* pbcCr = (const float*)d_in[3];
    const float* pbcCi = (const float*)d_in[4];
    const float* C1r   = (const float*)d_in[5];
    const float* C1i   = (const float*)d_in[6];
    const float* C2r   = (const float*)d_in[7];
    const float* C2i   = (const float*)d_in[8];

    int2* mn = (int2*)d_ws;
    unsigned int* cpack = (unsigned int*)((char*)d_ws + WS_CPACK_OFF);

    build_mn<<<1, 32, 0, stream>>>(mn);

    const int total = NB * 4 * NCHUNK * 32 * 8;           // 30720 dwords
    pack_coefs<<<(total + 255) / 256, 256, 0, stream>>>(
        task, pbcCr, pbcCi, C1r, C1i, C2r, C2i, cpack);

    sopbc_main<<<NB * (T_LEN / TT), 256, 0, stream>>>(
        Er, Ei, task, mn, cpack, (float*)d_out);
}
